// TrunkJ_65807488910116
// MI455X (gfx1250) — matrix operand order repaired and checked by reading, not compiled, not run on hardware
//
#include <hip/hip_runtime.h>
#include <hip/hip_bf16.h>

// ---------------------------------------------------------------------------
// Types for CDNA5 WMMA (wave32, 16x16x32 bf16 -> f32)
// ---------------------------------------------------------------------------
typedef __attribute__((ext_vector_type(8)))  __bf16 v8bf;
typedef __attribute__((ext_vector_type(16))) __bf16 v16bf;
typedef __attribute__((ext_vector_type(8)))  float  v8f;

constexpr int Bn = 32, C1 = 128, Hh = 32, Ww = 32, Dd = 256, NHh = 8, Ss = 1024;
constexpr int KTOP = 66;

__device__ inline v8f wmma_bf16(v16bf a, v16bf b, v8f c) {
  return __builtin_amdgcn_wmma_f32_16x16x32_bf16(false, a, false, b, (short)0, c,
                                                 false, false);
}

// A-matrix 16x32 bf16 fragment from row-major A (lda elements per row).
// ISA layout: lane<16 -> row=lane, K chunks {0..7,16..23}; lane>=16 -> row=lane-16,
// chunks {8..15,24..31}.
__device__ inline v16bf load_A16x32(const __bf16* A, int lda, int m0, int k0, int lane) {
  int half = lane >> 4, r = lane & 15;
  const __bf16* base = A + (size_t)(m0 + r) * lda + k0 + half * 8;
  v8bf c0 = *(const v8bf*)(base);
  v8bf c1 = *(const v8bf*)(base + 16);
  v16bf a;
#pragma unroll
  for (int i = 0; i < 8; ++i) { a[i] = c0[i]; a[8 + i] = c1[i]; }
  return a;
}

// B-matrix 32x16 bf16 fragment from N-major Bt (row = N, ldb = K stride).
// ISA layout: lane<16 -> col N=lane, K chunks {0..7,16..23}; lane>=16 -> col N=lane-16,
// chunks {8..15,24..31}.
__device__ inline v16bf load_Bt32x16(const __bf16* Bt, int ldb, int n0, int k0, int lane) {
  int half = lane >> 4, c = lane & 15;
  const __bf16* base = Bt + (size_t)(n0 + c) * ldb + k0 + half * 8;
  v8bf c0 = *(const v8bf*)(base);
  v8bf c1 = *(const v8bf*)(base + 16);
  v16bf b;
#pragma unroll
  for (int i = 0; i < 8; ++i) { b[i] = c0[i]; b[8 + i] = c1[i]; }
  return b;
}

// ---------------------------------------------------------------------------
// Stage 1: conv(1->128, 9x9, pad 4) + ReLU
// ---------------------------------------------------------------------------
__global__ void k_conv_relu(const float* __restrict__ x, const float* __restrict__ cw,
                            float* __restrict__ A) {
  int idx = blockIdx.x * blockDim.x + threadIdx.x;
  if (idx >= Bn * C1 * Hh * Ww) return;
  int w = idx & 31, h = (idx >> 5) & 31, c = (idx >> 10) & 127, b = idx >> 17;
  const float* xb = x + (size_t)b * (Hh * Ww);
  const float* wc = cw + c * 81;
  float acc = 0.f;
#pragma unroll
  for (int i = 0; i < 9; ++i) {
    int hy = h + i - 4;
    if (hy < 0 || hy >= Hh) continue;
#pragma unroll
    for (int j = 0; j < 9; ++j) {
      int wx = w + j - 4;
      if (wx < 0 || wx >= Ww) continue;
      acc += xb[hy * Ww + wx] * wc[i * 9 + j];
    }
  }
  A[idx] = acc > 0.f ? acc : 0.f;
}

// ---------------------------------------------------------------------------
// Stage 2: per-batch global max + exact top-k threshold (binary search on
// float bit pattern; values are >= 0 after ReLU so uint compare is monotonic)
// ---------------------------------------------------------------------------
__global__ void k_thresh(const float* __restrict__ A, float* __restrict__ thr,
                         float* __restrict__ gmax) {
  int b = blockIdx.x, tid = threadIdx.x;
  const float* Ab = A + (size_t)b * (C1 * Hh * Ww);
  __shared__ float sred[256];
  __shared__ int scnt;
  float mx = 0.f;
  for (int i = tid; i < C1 * Hh * Ww; i += 256) mx = fmaxf(mx, Ab[i]);
  sred[tid] = mx;
  __syncthreads();
  for (int s = 128; s > 0; s >>= 1) {
    if (tid < s) sred[tid] = fmaxf(sred[tid], sred[tid + s]);
    __syncthreads();
  }
  float gm = sred[0];
  __syncthreads();
  unsigned lo = 0, hi = __float_as_uint(gm);
  while (lo < hi) {
    unsigned mid = lo + ((hi - lo + 1) >> 1);
    if (tid == 0) scnt = 0;
    __syncthreads();
    int cnt = 0;
    for (int i = tid; i < C1 * Hh * Ww; i += 256)
      cnt += (__float_as_uint(Ab[i]) >= mid) ? 1 : 0;
    atomicAdd(&scnt, cnt);
    __syncthreads();
    int total = scnt;
    __syncthreads();
    if (total >= KTOP) lo = mid; else hi = mid - 1;
  }
  if (tid == 0) { thr[b] = __uint_as_float(lo); gmax[b] = gm; }
}

// ---------------------------------------------------------------------------
// Stage 3: per-pixel channel top-8 + scatter to (B,H,W,C) logits; emit topi
// ---------------------------------------------------------------------------
__global__ void k_topk(const float* __restrict__ A, const float* __restrict__ thr,
                       float* __restrict__ logits, int* __restrict__ topi) {
  int idx = blockIdx.x * blockDim.x + threadIdx.x;  // pixel id (b*1024 + h*32 + w)
  if (idx >= Bn * Hh * Ww) return;
  int b = idx >> 10, pix = idx & 1023;
  const float* Ap = A + (size_t)b * (C1 * Hh * Ww) + pix;
  float th = thr[b];
  float v[8];
  int id[8];
#pragma unroll
  for (int j = 0; j < 8; ++j) { v[j] = -1.f; id[j] = 0; }
  for (int c = 0; c < C1; ++c) {
    float a = Ap[(size_t)c * (Hh * Ww)];
    float val = (a >= th) ? a : 0.f;
    if (val > v[7]) {  // strict > keeps lowest index first on ties (jax semantics)
      v[7] = val; id[7] = c;
#pragma unroll
      for (int j = 7; j > 0; --j) {
        if (v[j] > v[j - 1]) {
          float tv = v[j]; v[j] = v[j - 1]; v[j - 1] = tv;
          int ti = id[j]; id[j] = id[j - 1]; id[j - 1] = ti;
        }
      }
    }
  }
  float* lp = logits + (size_t)idx * C1;
  for (int c = 0; c < C1; ++c) lp[c] = 0.f;
#pragma unroll
  for (int j = 0; j < 8; ++j) { lp[id[j]] = v[j]; topi[idx * 8 + j] = id[j]; }
}

// Stage 4: normalize by per-batch gmax (in place in d_out) + bf16 copy for WMMA
__global__ void k_norm(float* __restrict__ sw, const float* __restrict__ gmax,
                       __bf16* __restrict__ swb) {
  int i = blockIdx.x * blockDim.x + threadIdx.x;
  if (i >= Bn * Hh * Ww * C1) return;
  float g = gmax[i >> 17];  // 131072 elems per batch
  float o = (g == 0.f) ? 0.f : sw[i] / g;
  sw[i] = o;
  swb[i] = (__bf16)o;
}

// Positional encoding pe[w][d] (varies only along W, per reference broadcast)
__global__ void k_pe(float* __restrict__ pe) {
  int i = blockIdx.x * blockDim.x + threadIdx.x;
  if (i >= Ww * Dd) return;
  int w = i >> 8, n = i & 255;
  float dv = __expf((float)(n & ~1) * (-9.210340371976184f / (float)Dd));
  float ang = (float)w * dv;
  pe[i] = (n & 1) ? __cosf(ang) : __sinf(ang);
}

// Weight conversions
__global__ void k_f32_to_bf16(const float* __restrict__ s, __bf16* __restrict__ d, int n) {
  int i = blockIdx.x * blockDim.x + threadIdx.x;
  if (i < n) d[i] = (__bf16)s[i];
}
__global__ void k_ce_T(const float* __restrict__ ce, __bf16* __restrict__ ceT) {
  int i = blockIdx.x * blockDim.x + threadIdx.x;  // (n,k) over 256x128
  if (i >= Dd * C1) return;
  int n = i >> 7, k = i & 127;
  ceT[i] = (__bf16)ce[(size_t)k * Dd + n];
}
__global__ void k_init_pooled(float* __restrict__ pooled, const float* __restrict__ ob) {
  int i = blockIdx.x * blockDim.x + threadIdx.x;
  if (i < Bn * Dd) pooled[i] = ob[i & 255];
}

// ---------------------------------------------------------------------------
// Stage 5 (WMMA): seq = sparse_weights(32768x128) @ ce(128x256) + PE -> bf16
// ---------------------------------------------------------------------------
__global__ void __launch_bounds__(128) k_gemm_seq(const __bf16* __restrict__ Aw,
    const __bf16* __restrict__ Bt, const float* __restrict__ pe,
    __bf16* __restrict__ seq) {
  int wave = threadIdx.x >> 5, lane = threadIdx.x & 31;
  int tid = blockIdx.x * 4 + wave;  // tiles: 2048 (M/16) x 8 (N/32)
  int m0 = (tid >> 3) * 16, n0 = (tid & 7) * 32;
  v8f acc0 = {}, acc1 = {};
#pragma unroll
  for (int k = 0; k < C1; k += 32) {
    v16bf a = load_A16x32(Aw, C1, m0, k, lane);
    v16bf b0 = load_Bt32x16(Bt, C1, n0, k, lane);
    v16bf b1 = load_Bt32x16(Bt, C1, n0 + 16, k, lane);
    acc0 = wmma_bf16(a, b0, acc0);
    acc1 = wmma_bf16(a, b1, acc1);
  }
  int half = lane >> 4, col = lane & 15;
#pragma unroll
  for (int r = 0; r < 8; ++r) {
    int m = m0 + r + 8 * half, w = m & 31, n = n0 + col;
    seq[(size_t)m * Dd + n]      = (__bf16)(acc0[r] + pe[w * Dd + n]);
    seq[(size_t)m * Dd + n + 16] = (__bf16)(acc1[r] + pe[w * Dd + n + 16]);
  }
}

// ---------------------------------------------------------------------------
// Stage 6 (WMMA): qkv = seq(32768x256) @ Wqkv^T(256x768) + b, routed to
// q[b][h][s][32], k[b][h][s][32], vT[b][h][32][s] (V transposed for PV B-frags)
// ---------------------------------------------------------------------------
__global__ void __launch_bounds__(128) k_gemm_qkv(const __bf16* __restrict__ seq,
    const __bf16* __restrict__ Wt, const float* __restrict__ bias,
    __bf16* __restrict__ qb, __bf16* __restrict__ kb, __bf16* __restrict__ vT) {
  int wave = threadIdx.x >> 5, lane = threadIdx.x & 31;
  int tid = blockIdx.x * 4 + wave;  // tiles: 2048 x 24
  int m0 = (tid / 24) * 16, n0 = (tid % 24) * 32;
  v8f acc0 = {}, acc1 = {};
  for (int k = 0; k < Dd; k += 32) {
    v16bf a = load_A16x32(seq, Dd, m0, k, lane);
    v16bf b0 = load_Bt32x16(Wt, Dd, n0, k, lane);
    v16bf b1 = load_Bt32x16(Wt, Dd, n0 + 16, k, lane);
    acc0 = wmma_bf16(a, b0, acc0);
    acc1 = wmma_bf16(a, b1, acc1);
  }
  int half = lane >> 4, col = lane & 15;
#pragma unroll
  for (int r = 0; r < 8; ++r) {
    int m = m0 + r + 8 * half, b = m >> 10, s = m & 1023;
#pragma unroll
    for (int t = 0; t < 2; ++t) {
      int n = n0 + col + t * 16;
      __bf16 bv = (__bf16)((t ? acc1[r] : acc0[r]) + bias[n]);
      if (n < 256) {
        int h = n >> 5, d = n & 31;
        qb[(((size_t)(b * NHh + h)) * Ss + s) * 32 + d] = bv;
      } else if (n < 512) {
        int nn = n - 256, h = nn >> 5, d = nn & 31;
        kb[(((size_t)(b * NHh + h)) * Ss + s) * 32 + d] = bv;
      } else {
        int nn = n - 512, h = nn >> 5, d = nn & 31;
        vT[(((size_t)(b * NHh + h)) * 32 + d) * Ss + s] = bv;
      }
    }
  }
}

// ---------------------------------------------------------------------------
// Stage 7 (WMMA): flash attention. 1 wave = 16 q rows; dh=32 = one WMMA K.
// C-layout rows live per 16-lane half -> shfl_xor row reductions; P goes
// through per-wave LDS to relayout C->A between score and PV WMMAs.
// ---------------------------------------------------------------------------
__global__ void __launch_bounds__(128) k_attn(const __bf16* __restrict__ qbuf,
    const __bf16* __restrict__ kbuf, const __bf16* __restrict__ vT,
    __bf16* __restrict__ ctx) {
  __shared__ alignas(16) __bf16 Plds[4][16 * 32];
  int wave = threadIdx.x >> 5, lane = threadIdx.x & 31;
  int qblk = blockIdx.x & 15, bh = blockIdx.x >> 4;
  int b = bh >> 3, h = bh & 7;
  int q0 = qblk * 64 + wave * 16;
  const __bf16* qh = qbuf + ((size_t)(b * NHh + h)) * Ss * 32;
  const __bf16* kh = kbuf + ((size_t)(b * NHh + h)) * Ss * 32;
  const __bf16* vh = vT + ((size_t)(b * NHh + h)) * 32 * Ss;
  int half = lane >> 4, col = lane & 15;
  v16bf aq = load_A16x32(qh, 32, q0, 0, lane);
  v8f o0 = {}, o1 = {};
  float rm[8], rs[8];
#pragma unroll
  for (int r = 0; r < 8; ++r) { rm[r] = -1e30f; rs[r] = 0.f; }
  const float scale = 0.17677669529663687f;  // 1/sqrt(32)
  __bf16* Pw = &Plds[wave][0];
  for (int kt = 0; kt < Ss; kt += 32) {
    v16bf bk0 = load_Bt32x16(kh, 32, kt, 0, lane);       // keys kt..kt+15
    v16bf bk1 = load_Bt32x16(kh, 32, kt + 16, 0, lane);  // keys kt+16..kt+31
    v8f s0 = {}, s1 = {};
    s0 = wmma_bf16(aq, bk0, s0);
    s1 = wmma_bf16(aq, bk1, s1);
#pragma unroll
    for (int r = 0; r < 8; ++r) {
      float x0 = s0[r] * scale, x1 = s1[r] * scale;
      float mb = fmaxf(x0, x1);
#pragma unroll
      for (int msk = 1; msk < 16; msk <<= 1) mb = fmaxf(mb, __shfl_xor(mb, msk, 32));
      float mnew = fmaxf(rm[r], mb);
      float alpha = __expf(rm[r] - mnew);
      float p0 = __expf(x0 - mnew), p1 = __expf(x1 - mnew);
      float ls = p0 + p1;
#pragma unroll
      for (int msk = 1; msk < 16; msk <<= 1) ls += __shfl_xor(ls, msk, 32);
      rs[r] = rs[r] * alpha + ls;
      rm[r] = mnew;
      o0[r] *= alpha;
      o1[r] *= alpha;
      int row = 8 * half + r;
      Pw[row * 32 + col]      = (__bf16)p0;
      Pw[row * 32 + 16 + col] = (__bf16)p1;
    }
    asm volatile("s_wait_dscnt 0" ::: "memory");  // P tile visible wave-wide
    v16bf pa;
    {
      const __bf16* base = Pw + (size_t)col * 32 + half * 8;
      v8bf c0 = *(const v8bf*)(base);
      v8bf c1 = *(const v8bf*)(base + 16);
#pragma unroll
      for (int i = 0; i < 8; ++i) { pa[i] = c0[i]; pa[8 + i] = c1[i]; }
    }
    v16bf bv0 = load_Bt32x16(vh, Ss, 0, kt, lane);   // d = 0..15
    v16bf bv1 = load_Bt32x16(vh, Ss, 16, kt, lane);  // d = 16..31
    o0 = wmma_bf16(pa, bv0, o0);
    o1 = wmma_bf16(pa, bv1, o1);
  }
#pragma unroll
  for (int r = 0; r < 8; ++r) {
    float inv = 1.f / rs[r];
    int q = q0 + r + 8 * half;
    size_t base = ((size_t)(b * Ss + q)) * Dd + h * 32;
    ctx[base + col]      = (__bf16)(o0[r] * inv);
    ctx[base + col + 16] = (__bf16)(o1[r] * inv);
  }
}

// ---------------------------------------------------------------------------
// Stage 8 (WMMA): out-proj GEMM fused with mean pooling (column reduce +
// one atomicAdd per column per wave; pooled pre-initialized with bias)
// ---------------------------------------------------------------------------
__global__ void __launch_bounds__(128) k_gemm_out(const __bf16* __restrict__ ctx,
    const __bf16* __restrict__ Wt, float* __restrict__ pooled) {
  int wave = threadIdx.x >> 5, lane = threadIdx.x & 31;
  int tid = blockIdx.x * 4 + wave;  // tiles: 2048 x 8
  int m0 = (tid >> 3) * 16, n0 = (tid & 7) * 32;
  v8f acc0 = {}, acc1 = {};
  for (int k = 0; k < Dd; k += 32) {
    v16bf a = load_A16x32(ctx, Dd, m0, k, lane);
    v16bf b0 = load_Bt32x16(Wt, Dd, n0, k, lane);
    v16bf b1 = load_Bt32x16(Wt, Dd, n0 + 16, k, lane);
    acc0 = wmma_bf16(a, b0, acc0);
    acc1 = wmma_bf16(a, b1, acc1);
  }
  int bidx = m0 >> 10;
  float c0 = 0.f, c1 = 0.f;
#pragma unroll
  for (int r = 0; r < 8; ++r) { c0 += acc0[r]; c1 += acc1[r]; }
  c0 += __shfl_xor(c0, 16, 32);  // combine row halves (M 0-7 and 8-15)
  c1 += __shfl_xor(c1, 16, 32);
  if (lane < 16) {
    atomicAdd(pooled + bidx * Dd + n0 + lane,      c0 * (1.f / 1024.f));
    atomicAdd(pooled + bidx * Dd + n0 + 16 + lane, c1 * (1.f / 1024.f));
  }
}

// ---------------------------------------------------------------------------
extern "C" void kernel_launch(void* const* d_in, const int* in_sizes, int n_in,
                              void* d_out, int out_size, void* d_ws, size_t ws_size,
                              hipStream_t stream) {
  (void)in_sizes; (void)n_in; (void)out_size; (void)ws_size;
  const float* x    = (const float*)d_in[0];
  const float* cw   = (const float*)d_in[1];
  const float* ce   = (const float*)d_in[2];
  const float* wqkv = (const float*)d_in[3];
  const float* bqkv = (const float*)d_in[4];
  const float* wout = (const float*)d_in[5];
  const float* bout = (const float*)d_in[6];

  float* out    = (float*)d_out;
  float* pooled = out;                       // 32*256
  float* sw_out = out + 8192;                // 32*32*32*128 (B,H,W,C)
  int*   topi   = (int*)(out + 8192 + 4194304);

  char* w = (char*)d_ws;
  float*  A    = (float*)(w + 0);            // 16 MB  conv activations
  float*  thr  = (float*)(w + 16777216);     // 32 f32
  float*  gmax = (float*)(w + 16777344);     // 32 f32
  __bf16* swb  = (__bf16*)(w + 16777472);    // 8 MB   sparse weights bf16
  float*  pe   = (float*)(w + 25166080);     // 32 KB
  __bf16* ceT  = (__bf16*)(w + 25198848);    // 64 KB  ce transposed (N,K)
  __bf16* wqb  = (__bf16*)(w + 25264384);    // 384 KB in_proj_w bf16 (N,K)
  __bf16* wob  = (__bf16*)(w + 25657600);    // 128 KB out_proj_w bf16 (N,K)
  __bf16* seqb = (__bf16*)(w + 25788672);    // 16 MB  seq bf16
  __bf16* qbf  = (__bf16*)(w + 42565888);    // 16 MB  q[b][h][s][32]
  __bf16* kbf  = (__bf16*)(w + 59343104);    // 16 MB  k[b][h][s][32]
  __bf16* vTb  = (__bf16*)(w + 76120320);    // 16 MB  v[b][h][32][s]
  __bf16* ctxb = (__bf16*)(w + 92897536);    // 16 MB  ctx bf16

  k_conv_relu   <<<16384, 256, 0, stream>>>(x, cw, A);
  k_thresh      <<<   32, 256, 0, stream>>>(A, thr, gmax);
  k_topk        <<<  128, 256, 0, stream>>>(A, thr, sw_out, topi);
  k_norm        <<<16384, 256, 0, stream>>>(sw_out, gmax, swb);
  k_pe          <<<   32, 256, 0, stream>>>(pe);
  k_ce_T        <<<  128, 256, 0, stream>>>(ce, ceT);
  k_f32_to_bf16 <<<  768, 256, 0, stream>>>(wqkv, wqb, 196608);
  k_f32_to_bf16 <<<  256, 256, 0, stream>>>(wout, wob, 65536);
  k_init_pooled <<<   32, 256, 0, stream>>>(pooled, bout);
  k_gemm_seq    <<< 4096, 128, 0, stream>>>(swb, ceT, pe, seqb);
  k_gemm_qkv    <<<12288, 128, 0, stream>>>(seqb, wqb, bqkv, qbf, kbf, vTb);
  k_attn        <<< 4096, 128, 0, stream>>>(qbf, kbf, vTb, ctxb);
  k_gemm_out    <<< 4096, 128, 0, stream>>>(ctxb, wob, pooled);
}
